// MogrifierEDRN_51659866637125
// MI455X (gfx1250) — matrix operand order repaired and checked by reading, not compiled, not run on hardware
//
#include <hip/hip_runtime.h>
#include <hip/hip_bf16.h>
#include <math.h>

// Problem constants (match reference)
#define Bb  64
#define Ss  512
#define Nn  512
#define Mm  1024
#define MDm 2048

typedef __bf16 bf16_t;
typedef __attribute__((ext_vector_type(16))) __bf16 v16bf;
typedef __attribute__((ext_vector_type(8)))  __bf16 v8bf;
typedef __attribute__((ext_vector_type(8)))  float  v8f;

union ABFrag { v16bf v; v8bf h[2]; };

#define LDS_TILE (32 * 64)   // one 32-row x 64-K bf16 A tile

struct EpiArgs {
  const float* bias;      // bm (mode 0) or b (mode 1)
  const float* x;         // input sequence (B,S,N)
  bf16_t* xm_bf;          // mogrified x, bf16 (B,N)
  float*  bprim;          // (B, 4*MD) running pre-activations
  const float* c_in_f32;  // c used by epilogue (old c for mode3, new c for mode4)
  float*  c_out_f32;
  bf16_t* c_out_bf;
  float*  cc_f32;
  bf16_t* cc_bf;
  float*  h_f32;
  bf16_t* h_bf;
  float*  out_seq;        // hidden_seq (B,S,M)
  int t;
};

__device__ __forceinline__ float sigm(float v) { return 1.0f / (1.0f + __expf(-v)); }

// Async copy of one 16B chunk: global -> LDS, tracked by ASYNCcnt (no VGPR landing).
__device__ __forceinline__ void async_copy_b128(unsigned lds_byte_addr,
                                                unsigned long long gaddr) {
  asm volatile("global_load_async_to_lds_b128 %0, %1, off"
               :: "v"(lds_byte_addr), "v"(gaddr) : "memory");
}

// One GEMM pass: acc[16x32 per wave] += A(32xK tile rows) x Bt(N-major weights)
// A is bf16 row-major (lda), Bt is bf16 N-major: Bt[n*K + k].
// A tiles are double-buffered in LDS via async-to-LDS copies.
__device__ __forceinline__ void gemm_pass(const bf16_t* __restrict__ A, int lda,
                                          const bf16_t* __restrict__ Bt, int K,
                                          int col0, bf16_t* lds_a, /* [2][32*64] */
                                          v8f& acc0, v8f& acc1)
{
  const int tid  = threadIdx.x;
  const int lane = tid & 31;
  const int wave = tid >> 5;
  const int wr   = wave >> 2;          // 0..1  (row group of 16)
  const int half = lane >> 4;          // 0/1
  const int r    = lane & 15;
  const int ldrow   = tid >> 3;        // 0..31 : cooperative A-tile copy row
  const int ldchunk = (tid & 7) << 3;  // 0..56 step 8 (one 16B chunk per thread)
  const int wrow = wr << 4;

  const bf16_t* gsrc = A + (size_t)ldrow * lda + ldchunk;
  const unsigned ldst0 = (unsigned)(uintptr_t)(lds_a + ldrow * 64 + ldchunk);
  const unsigned ldst1 = ldst0 + LDS_TILE * 2;   // second buffer (bytes)

  // prologue: tile 0 -> buffer 0
  async_copy_b128(ldst0, (unsigned long long)(uintptr_t)gsrc);

  const int T = K >> 6;                // K / 64 tiles
  for (int i = 0; i < T; ++i) {
    if (i + 1 < T) {
      // overlap: async-copy tile i+1 into the other buffer while we compute
      async_copy_b128((i & 1) ? ldst0 : ldst1,
                      (unsigned long long)(uintptr_t)(gsrc + (size_t)(i + 1) * 64));
      asm volatile("s_wait_asynccnt 0x1" ::: "memory");  // tile i resident
    } else {
      asm volatile("s_wait_asynccnt 0x0" ::: "memory");
    }
    __syncthreads();                   // tile i visible to all waves

    const bf16_t* buf = lds_a + ((i & 1) ? LDS_TILE : 0);
    const int k0 = i << 6;
#pragma unroll
    for (int s = 0; s < 2; ++s) {
      // A fragment (16x32 bf16): lane<16 -> K {0..15}, lane>=16 -> {16..31}
      ABFrag af;
      const bf16_t* arow = buf + (wrow + r) * 64 + s * 32 + half * 16;
      af.h[0] = *(const v8bf*)(arow);
      af.h[1] = *(const v8bf*)(arow + 8);

      // B fragments (32x16 bf16): lane holds column n = col0 + j*16 + r,
      // K = half*16 + 0..15 -> 16 contiguous bf16 in N-major weights
      const int kk = k0 + s * 32 + half * 16;
      ABFrag b0, b1;
      const bf16_t* bp0 = Bt + (size_t)(col0 + r) * K + kk;
      b0.h[0] = *(const v8bf*)bp0;
      b0.h[1] = *(const v8bf*)(bp0 + 8);
      const bf16_t* bp1 = Bt + (size_t)(col0 + 16 + r) * K + kk;
      b1.h[0] = *(const v8bf*)bp1;
      b1.h[1] = *(const v8bf*)(bp1 + 8);

      acc0 = __builtin_amdgcn_wmma_f32_16x16x32_bf16(false, af.v, false, b0.v,
                                                     (short)0, acc0, false, false);
      acc1 = __builtin_amdgcn_wmma_f32_16x16x32_bf16(false, af.v, false, b1.v,
                                                     (short)0, acc1, false, false);
    }
    __syncthreads();                   // done reading buf before it is overwritten
  }
}

template<int MODE>
__device__ __forceinline__ void epi(int row, int col, float acc, const EpiArgs& ep)
{
  if (MODE == 0) {            // x_m = 2*sigmoid(h@Wm.T + bm) * x_t   -> bf16
    float v = 2.0f * sigm(acc + ep.bias[col]) *
              ep.x[((size_t)row * Ss + ep.t) * Nn + col];
    ep.xm_bf[row * Nn + col] = (bf16_t)v;
  } else if (MODE == 1) {     // b_prim = x_m@Bw + b
    ep.bprim[(size_t)row * 4 * MDm + col] = acc + ep.bias[col];
  } else if (MODE == 2) {     // a_prim = h@A_fg_in + b_prim[:, :2MD] (in place)
    float* p = ep.bprim + (size_t)row * 4 * MDm + col;
    *p += acc;
  } else if (MODE == 3) {     // cell update (acc = c@A_pt + c_tail@A_th)
    const float* bp = ep.bprim + (size_t)row * 4 * MDm;
    float gth = tanhf(acc + bp[2 * MDm + col]);
    float gfg = sigm(bp[col]);
    float gin = sigm(bp[MDm + col]);
    float cn  = ep.c_in_f32[row * MDm + col] * gfg + gth * gin;
    ep.c_out_f32[row * MDm + col] = cn;
    ep.c_out_bf[row * MDm + col]  = (bf16_t)cn;
  } else if (MODE == 4) {     // cc = tanh(c_new)*sigmoid(c_tail@A_ot + b_prim[3MD:])
    const float* bp = ep.bprim + (size_t)row * 4 * MDm;
    float got = sigm(acc + bp[3 * MDm + col]);
    float cc  = tanhf(ep.c_in_f32[row * MDm + col]) * got;
    ep.cc_f32[row * MDm + col] = cc;
    ep.cc_bf[row * MDm + col]  = (bf16_t)cc;
  } else {                    // h = cc_tail@A_st + cc[:, 0:M]; write hidden_seq
    float hv = acc + ep.cc_f32[row * MDm + col];
    ep.h_f32[row * Mm + col] = hv;
    ep.h_bf[row * Mm + col]  = (bf16_t)hv;
    ep.out_seq[((size_t)row * Ss + ep.t) * Mm + col] = hv;
  }
}

// 256 threads = 8 wave32; wave tile 16x32 (2 WMMA accumulators), WG tile 32x128.
// grid = (Ncols/128, 64/32)
template<int MODE>
__global__ __launch_bounds__(256) void gemm_rnn(
    const bf16_t* __restrict__ A, int lda,
    const bf16_t* __restrict__ Bt, int K,
    const bf16_t* __restrict__ A2,
    const bf16_t* __restrict__ B2t, int K2,
    EpiArgs ep)
{
  __shared__ __align__(16) bf16_t lds_a[2 * LDS_TILE];   // double-buffered A tile
  const int tid  = threadIdx.x;
  const int lane = tid & 31;
  const int wave = tid >> 5;
  const int wr   = wave >> 2, wc = wave & 3;
  const int row0 = blockIdx.y * 32;
  const int wcol0 = blockIdx.x * 128 + wc * 32;

  v8f acc0 = {}; v8f acc1 = {};
  gemm_pass(A + (size_t)row0 * lda, lda, Bt, K, wcol0, lds_a, acc0, acc1);
  if (MODE == 3 && K2 > 0)   // second K-segment: c_tail @ A_th
    gemm_pass(A2 + (size_t)row0 * lda, lda, B2t, K2, wcol0, lds_a, acc0, acc1);

  const int r = lane & 15, half = lane >> 4;
  const int gr0 = row0 + wr * 16 + half * 8;
#pragma unroll
  for (int i = 0; i < 8; ++i) {
    epi<MODE>(gr0 + i, wcol0 + r,      acc0[i], ep);
    epi<MODE>(gr0 + i, wcol0 + 16 + r, acc1[i], ep);
  }
}

// ---- one-time weight conversion ----
__global__ void k_cvt(const float* __restrict__ in, bf16_t* __restrict__ out, int n) {
  int i = blockIdx.x * 256 + threadIdx.x;
  if (i < n) out[i] = (bf16_t)in[i];
}
// in: (K rows, Ncol cols) row-major -> out N-major: out[n*K + k]
__global__ void k_cvt_t(const float* __restrict__ in, bf16_t* __restrict__ out,
                        int K, int Ncol) {
  int i = blockIdx.x * 256 + threadIdx.x;
  if (i < K * Ncol) {
    int k = i / Ncol, n = i % Ncol;
    out[(size_t)n * K + k] = (bf16_t)in[i];
  }
}

__global__ void k_zero(float* hf, bf16_t* hb, float* cf, bf16_t* cb) {
  int i = blockIdx.x * 256 + threadIdx.x;
  if (i < Bb * Mm)  { hf[i] = 0.0f; hb[i] = (bf16_t)0.0f; }
  if (i < Bb * MDm) { cf[i] = 0.0f; cb[i] = (bf16_t)0.0f; }
}

__global__ void k_final(const float* __restrict__ hf, const float* __restrict__ cf,
                        float* __restrict__ out) {
  int i = blockIdx.x * 256 + threadIdx.x;
  const size_t base = (size_t)Bb * Ss * Mm;
  if (i < Bb * Mm)  out[base + i] = hf[i];
  if (i < Bb * MDm) out[base + (size_t)Bb * Mm + i] = cf[i];
}

extern "C" void kernel_launch(void* const* d_in, const int* in_sizes, int n_in,
                              void* d_out, int out_size, void* d_ws, size_t ws_size,
                              hipStream_t stream)
{
  (void)in_sizes; (void)n_in; (void)out_size; (void)ws_size;
  const float* x    = (const float*)d_in[0];
  const float* Wm   = (const float*)d_in[1];
  const float* bm   = (const float*)d_in[2];
  const float* Ath  = (const float*)d_in[3];
  const float* Aot  = (const float*)d_in[4];
  const float* Afg  = (const float*)d_in[5];
  const float* Apt  = (const float*)d_in[6];
  const float* Ast  = (const float*)d_in[7];
  const float* Bw   = (const float*)d_in[8];
  const float* bvec = (const float*)d_in[9];
  float* out = (float*)d_out;

  char* w = (char*)d_ws;
  size_t off = 0;
  auto alloc = [&](size_t bytes) -> char* {
    char* p = w + off; off += (bytes + 255) & ~(size_t)255; return p;
  };
  bf16_t* Wm_bf = (bf16_t*)alloc((size_t)Nn * Mm * 2);        // already N-major
  bf16_t* Ath_t = (bf16_t*)alloc((size_t)Mm * MDm * 2);
  bf16_t* Aot_t = (bf16_t*)alloc((size_t)Mm * MDm * 2);
  bf16_t* Afg_t = (bf16_t*)alloc((size_t)Mm * 2 * MDm * 2);
  bf16_t* Apt_t = (bf16_t*)alloc((size_t)MDm * MDm * 2);
  bf16_t* Ast_t = (bf16_t*)alloc((size_t)Mm * Mm * 2);
  bf16_t* Bw_t  = (bf16_t*)alloc((size_t)Nn * 4 * MDm * 2);
  float*  h_f32 = (float*) alloc((size_t)Bb * Mm * 4);
  bf16_t* h_bf  = (bf16_t*)alloc((size_t)Bb * Mm * 2);
  float*  cA_f  = (float*) alloc((size_t)Bb * MDm * 4);
  bf16_t* cA_b  = (bf16_t*)alloc((size_t)Bb * MDm * 2);
  float*  cB_f  = (float*) alloc((size_t)Bb * MDm * 4);
  bf16_t* cB_b  = (bf16_t*)alloc((size_t)Bb * MDm * 2);
  bf16_t* xm_bf = (bf16_t*)alloc((size_t)Bb * Nn * 2);
  float*  bprim = (float*) alloc((size_t)Bb * 4 * MDm * 4);
  float*  cc_f  = (float*) alloc((size_t)Bb * MDm * 4);
  bf16_t* cc_b  = (bf16_t*)alloc((size_t)Bb * MDm * 2);

  // one-time: pack weights to bf16, B-operand (N-major) layout
  k_cvt  <<<(Nn * Mm + 255) / 256, 256, 0, stream>>>(Wm, Wm_bf, Nn * Mm);
  k_cvt_t<<<(Mm * MDm + 255) / 256, 256, 0, stream>>>(Ath, Ath_t, Mm, MDm);
  k_cvt_t<<<(Mm * MDm + 255) / 256, 256, 0, stream>>>(Aot, Aot_t, Mm, MDm);
  k_cvt_t<<<(Mm * 2 * MDm + 255) / 256, 256, 0, stream>>>(Afg, Afg_t, Mm, 2 * MDm);
  k_cvt_t<<<(MDm * MDm + 255) / 256, 256, 0, stream>>>(Apt, Apt_t, MDm, MDm);
  k_cvt_t<<<(Mm * Mm + 255) / 256, 256, 0, stream>>>(Ast, Ast_t, Mm, Mm);
  k_cvt_t<<<(Nn * 4 * MDm + 255) / 256, 256, 0, stream>>>(Bw, Bw_t, Nn, 4 * MDm);
  k_zero<<<(Bb * MDm + 255) / 256, 256, 0, stream>>>(h_f32, h_bf, cA_f, cA_b);

  dim3 blk(256);
  for (int t = 0; t < Ss; ++t) {
    float*  c_in_f  = (t & 1) ? cB_f : cA_f;
    bf16_t* c_in_b  = (t & 1) ? cB_b : cA_b;
    float*  c_out_f = (t & 1) ? cA_f : cB_f;
    bf16_t* c_out_b = (t & 1) ? cA_b : cB_b;

    EpiArgs ep{};
    // 1) x_m = 2*sigmoid(h @ Wm.T + bm) * x_t      (64x512, K=1024)
    ep = EpiArgs{}; ep.t = t; ep.bias = bm; ep.x = x; ep.xm_bf = xm_bf;
    gemm_rnn<0><<<dim3(Nn / 128, 2), blk, 0, stream>>>(h_bf, Mm, Wm_bf, Mm,
                                                       nullptr, nullptr, 0, ep);
    // 2) b_prim = x_m @ Bw + b                     (64x8192, K=512)
    ep = EpiArgs{}; ep.t = t; ep.bias = bvec; ep.bprim = bprim;
    gemm_rnn<1><<<dim3((4 * MDm) / 128, 2), blk, 0, stream>>>(xm_bf, Nn, Bw_t, Nn,
                                                              nullptr, nullptr, 0, ep);
    // 3) b_prim[:, :2MD] += h @ A_fg_in            (64x4096, K=1024)
    ep = EpiArgs{}; ep.t = t; ep.bprim = bprim;
    gemm_rnn<2><<<dim3((2 * MDm) / 128, 2), blk, 0, stream>>>(h_bf, Mm, Afg_t, Mm,
                                                              nullptr, nullptr, 0, ep);
    // 4) c_new = c*sig(fg) + tanh(c@A_pt + c_tail@A_th + b_prim[2MD:])*sig(in)
    ep = EpiArgs{}; ep.t = t; ep.bprim = bprim;
    ep.c_in_f32 = c_in_f; ep.c_out_f32 = c_out_f; ep.c_out_bf = c_out_b;
    gemm_rnn<3><<<dim3(MDm / 128, 2), blk, 0, stream>>>(c_in_b, MDm, Apt_t, MDm,
                                                        c_in_b + Mm, Ath_t, Mm, ep);
    // 5) cc = tanh(c_new) * sigmoid(c_new_tail@A_ot + b_prim[3MD:])
    ep = EpiArgs{}; ep.t = t; ep.bprim = bprim;
    ep.c_in_f32 = c_out_f; ep.cc_f32 = cc_f; ep.cc_bf = cc_b;
    gemm_rnn<4><<<dim3(MDm / 128, 2), blk, 0, stream>>>(c_out_b + Mm, MDm, Aot_t, Mm,
                                                        nullptr, nullptr, 0, ep);
    // 6) h = cc_tail@A_st + cc[:, 0:M]; store hidden_seq[:, t, :]
    ep = EpiArgs{}; ep.t = t; ep.cc_f32 = cc_f;
    ep.h_f32 = h_f32; ep.h_bf = h_bf; ep.out_seq = out;
    gemm_rnn<5><<<dim3(Mm / 128, 2), blk, 0, stream>>>(cc_b + Mm, MDm, Ast_t, Mm,
                                                       nullptr, nullptr, 0, ep);
  }

  float* c_fin = ((Ss - 1) & 1) ? cA_f : cB_f;   // last step wrote this one
  k_final<<<(Bb * MDm + 255) / 256, 256, 0, stream>>>(h_f32, c_fin, out);
}
